// BeamSearchDecoder_79714593014242
// MI455X (gfx1250) — matrix operand order repaired and checked by reading, not compiled, not run on hardware
//
#include <hip/hip_runtime.h>

typedef __bf16 bf16_t;
typedef __attribute__((ext_vector_type(16))) __bf16 v16bf;
typedef __attribute__((ext_vector_type(8)))  float  v8f;

#define V_    50257
#define H_    512
#define S_    128
#define BEAM_ 5
#define TMAX_ 32
#define SOS_  1
#define EOS_  2
#define NEG_  (-1e9f)
#define NT_   3142            /* ceil(V/16) vocab tiles */
#define VP_   (NT_ * 16)      /* 50272 padded vocab */

/* beam-state layout (int32 words) */
#define ST_TOK 0
#define ST_SC  5
#define ST_FIN 10
#define ST_LEN 15
#define ST_SEQ 20
#define ST_SIZE 180

__device__ __forceinline__ float sigm_(float x) { return 1.0f / (1.0f + expf(-x)); }

/* ------------------------------------------------------------------ */
/* 1. source embedding lookup; also reset the global barrier counter  */
/* ------------------------------------------------------------------ */
__global__ void enc_embed_k(const int* __restrict__ toks,
                            const float* __restrict__ emb,
                            float* __restrict__ src_emb,
                            unsigned* __restrict__ ctr) {
    int idx = blockIdx.x * blockDim.x + threadIdx.x;
    if (idx == 0) *ctr = 0u;
    if (idx < S_ * H_) {
        int s = idx >> 9, j = idx & (H_ - 1);
        src_emb[idx] = emb[(long)toks[s] * H_ + j];
    }
}

/* ------------------------------------------------------------------ */
/* 2. GI[s][r] = dot(src_emb[s], Wih[r]) + bih[r]  (all steps, parallel) */
/* ------------------------------------------------------------------ */
__global__ void enc_gi_k(const float* __restrict__ src_emb,
                         const float* __restrict__ Wih,
                         const float* __restrict__ bih,
                         float* __restrict__ GI) {
    int idx = blockIdx.x * blockDim.x + threadIdx.x;
    if (idx >= S_ * 3 * H_) return;
    int s = idx / (3 * H_), r = idx % (3 * H_);
    const float4* x = (const float4*)(src_emb + s * H_);
    const float4* w = (const float4*)(Wih + (long)r * H_);
    float acc = 0.f;
    for (int k = 0; k < H_ / 4; ++k) {
        float4 a = x[k], b = w[k];
        acc += a.x * b.x + a.y * b.y + a.z * b.z + a.w * b.w;
    }
    GI[(long)s * 3 * H_ + r] = acc + bih[r];
}

/* ------------------------------------------------------------------ */
/* 3. pack W_vocab (f32 [H,V]) -> bf16 WMMA B-fragments               */
/*    layout: [tile][kc][lane][e]; lane<16: col=tile*16+lane, K=kc*32+e */
/*            lane>=16: col=tile*16+lane-16, K=kc*32+16+e             */
/* ------------------------------------------------------------------ */
__global__ void pack_wvocab_k(const float* __restrict__ Wv, bf16_t* __restrict__ Bp) {
    long idx = (long)blockIdx.x * blockDim.x + threadIdx.x;
    const long total = (long)NT_ * 16 * 32 * 16;
    if (idx >= total) return;
    int e = idx & 15;
    int lane = (idx >> 4) & 31;
    int kc = (idx >> 9) & 15;
    long tile = idx >> 13;
    int col, K;
    if (lane < 16) { col = (int)tile * 16 + lane;      K = kc * 32 + e; }
    else           { col = (int)tile * 16 + lane - 16; K = kc * 32 + 16 + e; }
    float v = (col < V_) ? Wv[(long)K * V_ + col] : 0.0f;
    Bp[idx] = (bf16_t)v;
}

/* ------------------------------------------------------------------ */
/* 4. persistent encoder scan: 16 WGs, global barrier per step        */
/* ------------------------------------------------------------------ */
__global__ void enc_scan_k(const float* __restrict__ GI,
                           const float* __restrict__ Whh,
                           const float* __restrict__ bhh,
                           float* __restrict__ enc_out,
                           unsigned* __restrict__ ctr) {
    __shared__ float hsh[H_];
    __shared__ float red[3 * 256];
    const int tid = threadIdx.x;
    const int w = blockIdx.x;               /* 0..15 */
    const int out = w * 32 + (tid >> 3);    /* h index this thread helps */
    const int sub = tid & 7;                /* 8 threads per output */
    const int k0 = sub * 64;
    volatile unsigned* vc = ctr;

    for (int s = 0; s < S_; ++s) {
        for (int k = tid; k < H_; k += 256)
            hsh[k] = (s == 0) ? 0.0f : enc_out[(long)(s - 1) * H_ + k];
        __syncthreads();

        float pr = 0.f, pz = 0.f, pn = 0.f;
        const float* wr = Whh + (long)out * H_;
        const float* wz = Whh + (long)(H_ + out) * H_;
        const float* wn = Whh + (long)(2 * H_ + out) * H_;
        for (int k = k0; k < k0 + 64; ++k) {
            float h = hsh[k];
            pr += h * wr[k]; pz += h * wz[k]; pn += h * wn[k];
        }
        red[tid] = pr; red[256 + tid] = pz; red[512 + tid] = pn;
        __syncthreads();
        if (sub == 0) {
            int base = tid;
            float gr = 0.f, gz = 0.f, gn = 0.f;
            for (int q = 0; q < 8; ++q) {
                gr += red[base + q]; gz += red[256 + base + q]; gn += red[512 + base + q];
            }
            int j = out;
            gr += bhh[j]; gz += bhh[H_ + j]; gn += bhh[2 * H_ + j];
            float ir = GI[(long)s * 3 * H_ + j];
            float iz = GI[(long)s * 3 * H_ + H_ + j];
            float in = GI[(long)s * 3 * H_ + 2 * H_ + j];
            float r = sigm_(ir + gr);
            float z = sigm_(iz + gz);
            float n = tanhf(in + r * gn);
            enc_out[(long)s * H_ + j] = (1.0f - z) * n + z * hsh[j];
        }
        __threadfence();
        __syncthreads();
        if (tid == 0) {
            atomicAdd(ctr, 1u);
            while (*vc < 16u * (unsigned)(s + 1)) __builtin_amdgcn_s_sleep(1);
        }
        __syncthreads();
        __threadfence();
    }
}

/* ------------------------------------------------------------------ */
/* 5. init beam state + hidden + zero A-pack rows                     */
/* ------------------------------------------------------------------ */
__global__ void init_state_k(float* __restrict__ hidden,
                             const float* __restrict__ enc_out,
                             int* __restrict__ st0,
                             bf16_t* __restrict__ Ap) {
    int tid = threadIdx.x;
    for (int i = tid; i < BEAM_ * H_; i += 256)
        hidden[i] = enc_out[(long)(S_ - 1) * H_ + (i & (H_ - 1))];
    for (int i = tid; i < 16 * 32 * 16; i += 256) Ap[i] = (bf16_t)0.0f;
    if (tid < BEAM_) {
        st0[ST_TOK + tid] = SOS_;
        ((float*)st0)[ST_SC + tid] = (tid == 0) ? 0.0f : NEG_;
        st0[ST_FIN + tid] = 0;
        st0[ST_LEN + tid] = 1;
    }
    for (int i = tid; i < BEAM_ * TMAX_; i += 256) st0[ST_SEQ + i] = 0;
}

/* ------------------------------------------------------------------ */
/* 6. decoder GRU: h2buf[b] = GRUCell(emb[tok_b], hidden[b])          */
/* ------------------------------------------------------------------ */
__global__ void dec_gru_k(const int* __restrict__ st_cur,
                          const float* __restrict__ emb,
                          const float* __restrict__ hidden,
                          const float* __restrict__ Wih,
                          const float* __restrict__ bih,
                          const float* __restrict__ Whh,
                          const float* __restrict__ bhh,
                          float* __restrict__ h2buf) {
    __shared__ float xs[H_], hs[H_];
    const int tid = threadIdx.x;
    const int b = blockIdx.x >> 1;
    const int j = ((blockIdx.x & 1) << 8) + tid;
    const int tok = st_cur[ST_TOK + b];
    for (int k = tid; k < H_; k += 256) {
        xs[k] = emb[(long)tok * H_ + k];
        hs[k] = hidden[(long)b * H_ + k];
    }
    __syncthreads();
    const float4* x4 = (const float4*)xs;
    const float4* h4 = (const float4*)hs;
    const float4* wir = (const float4*)(Wih + (long)j * H_);
    const float4* wiz = (const float4*)(Wih + (long)(H_ + j) * H_);
    const float4* win = (const float4*)(Wih + (long)(2 * H_ + j) * H_);
    const float4* whr = (const float4*)(Whh + (long)j * H_);
    const float4* whz = (const float4*)(Whh + (long)(H_ + j) * H_);
    const float4* whn = (const float4*)(Whh + (long)(2 * H_ + j) * H_);
    float ir = 0, iz = 0, in = 0, hr = 0, hz = 0, hn = 0;
    for (int k = 0; k < H_ / 4; ++k) {
        float4 xv = x4[k], hv = h4[k];
        float4 a;
        a = wir[k]; ir += xv.x * a.x + xv.y * a.y + xv.z * a.z + xv.w * a.w;
        a = wiz[k]; iz += xv.x * a.x + xv.y * a.y + xv.z * a.z + xv.w * a.w;
        a = win[k]; in += xv.x * a.x + xv.y * a.y + xv.z * a.z + xv.w * a.w;
        a = whr[k]; hr += hv.x * a.x + hv.y * a.y + hv.z * a.z + hv.w * a.w;
        a = whz[k]; hz += hv.x * a.x + hv.y * a.y + hv.z * a.z + hv.w * a.w;
        a = whn[k]; hn += hv.x * a.x + hv.y * a.y + hv.z * a.z + hv.w * a.w;
    }
    ir += bih[j]; iz += bih[H_ + j]; in += bih[2 * H_ + j];
    hr += bhh[j]; hz += bhh[H_ + j]; hn += bhh[2 * H_ + j];
    float r = sigm_(ir + hr);
    float z = sigm_(iz + hz);
    float n = tanhf(in + r * hn);
    h2buf[(long)b * H_ + j] = (1.0f - z) * n + z * hs[j];
}

/* ------------------------------------------------------------------ */
/* 7. attention + concat projection + A-fragment pack (one WG / beam) */
/* ------------------------------------------------------------------ */
__global__ void dec_attn_out_k(const float* __restrict__ h2buf,
                               const float* __restrict__ enc_out,
                               const float* __restrict__ Wc,
                               float* __restrict__ obuf,
                               bf16_t* __restrict__ Ap) {
    __shared__ float hs[H_], attn[S_], ctx[H_], os[H_], red[256];
    const int tid = threadIdx.x;
    const int b = blockIdx.x;
    for (int k = tid; k < H_; k += 256) hs[k] = h2buf[(long)b * H_ + k];
    __syncthreads();
    /* scores */
    if (tid < S_) {
        const float4* e4 = (const float4*)(enc_out + (long)tid * H_);
        const float4* h4 = (const float4*)hs;
        float acc = 0.f;
        for (int k = 0; k < H_ / 4; ++k) {
            float4 a = h4[k], c = e4[k];
            acc += a.x * c.x + a.y * c.y + a.z * c.z + a.w * c.w;
        }
        attn[tid] = acc;
    }
    __syncthreads();
    /* softmax over 128 */
    red[tid] = (tid < S_) ? attn[tid] : -INFINITY;
    __syncthreads();
    for (int off = 128; off > 0; off >>= 1) {
        if (tid < off) red[tid] = fmaxf(red[tid], red[tid + off]);
        __syncthreads();
    }
    float m = red[0];
    __syncthreads();
    if (tid < S_) attn[tid] = expf(attn[tid] - m);
    __syncthreads();
    red[tid] = (tid < S_) ? attn[tid] : 0.0f;
    __syncthreads();
    for (int off = 128; off > 0; off >>= 1) {
        if (tid < off) red[tid] += red[tid + off];
        __syncthreads();
    }
    float inv = 1.0f / red[0];
    __syncthreads();
    if (tid < S_) attn[tid] *= inv;
    __syncthreads();
    /* ctx */
    for (int j = tid; j < H_; j += 256) {
        float acc = 0.f;
        for (int i = 0; i < S_; ++i) acc += attn[i] * enc_out[(long)i * H_ + j];
        ctx[j] = acc;
    }
    __syncthreads();
    /* o = tanh(concat(h,ctx) @ Wc) */
    for (int j = tid; j < H_; j += 256) {
        float acc = 0.f;
        for (int k = 0; k < H_; ++k) acc += hs[k] * Wc[(long)k * H_ + j];
        for (int k = 0; k < H_; ++k) acc += ctx[k] * Wc[(long)(H_ + k) * H_ + j];
        float o = tanhf(acc);
        os[j] = o;
        obuf[(long)b * H_ + j] = o;
    }
    __syncthreads();
    /* pack this beam's row into the WMMA A fragment (contiguous k order, same as Bp) */
    for (int idx = tid; idx < 512; idx += 256) {
        int kc = idx >> 5, q = idx & 31;
        int hi = q >> 4, e = q & 15;
        int lane = b + (hi ? 16 : 0);
        int K = kc * 32 + e + (hi ? 16 : 0);
        Ap[((kc * 32 + lane) << 4) + e] = (bf16_t)os[K];
    }
}

/* ------------------------------------------------------------------ */
/* 8. logits via v_wmma_f32_16x16x32_bf16 — one vocab tile per wave   */
/* ------------------------------------------------------------------ */
__global__ void logits_wmma_k(const bf16_t* __restrict__ Ap,
                              const bf16_t* __restrict__ Bp,
                              float* __restrict__ logits) {
    const int lane = threadIdx.x & 31;
    const int wid = threadIdx.x >> 5;
    const long tile = (long)blockIdx.x * 8 + wid;
    if (tile >= NT_) return;                 /* wave-uniform */
    const bf16_t* bbase = Bp + tile * 8192;  /* 16 kc * 32 lane * 16 e */
    v8f c = {};
#pragma unroll
    for (int kc = 0; kc < 16; ++kc) {
        v16bf a = *(const v16bf*)(Ap + ((kc * 32 + lane) << 4));
        v16bf bm = *(const v16bf*)(bbase + kc * 512 + (lane << 4));
        if (kc + 1 < 16)
            __builtin_prefetch((const void*)(bbase + (kc + 1) * 512 + (lane << 4)), 0, 1);
        c = __builtin_amdgcn_wmma_f32_16x16x32_bf16(false, a, false, bm,
                                                    (short)0, c, false, false);
    }
    if (lane < 16) {
        long col = tile * 16 + lane;
#pragma unroll
        for (int r = 0; r < BEAM_; ++r)      /* rows 0..4 live in lanes 0-15 */
            logits[(long)r * VP_ + col] = c[r];
    }
}

/* ------------------------------------------------------------------ */
/* 9. logsumexp, two passes                                            */
/* ------------------------------------------------------------------ */
__global__ void lse_p1_k(const float* __restrict__ logits, float2* __restrict__ part) {
    __shared__ float sm[256], ss[256];
    const int tid = threadIdx.x;
    const int b = blockIdx.x >> 5;           /* 32 chunks per beam */
    const int c = blockIdx.x & 31;
    const int CH = (V_ + 31) / 32;           /* 1571 */
    const int v0 = c * CH, v1 = min(V_, v0 + CH);
    float m = -INFINITY, s = 0.0f;
    for (int v = v0 + tid; v < v1; v += 256) {
        float x = logits[(long)b * VP_ + v];
        if (x > m) { s = s * expf(m - x) + 1.0f; m = x; }
        else       { s += expf(x - m); }
    }
    sm[tid] = m; ss[tid] = s;
    __syncthreads();
    for (int off = 128; off > 0; off >>= 1) {
        if (tid < off) {
            float m2 = sm[tid + off], s2 = ss[tid + off];
            float m1 = sm[tid], s1 = ss[tid];
            if (m2 > m1) { float t = m1; m1 = m2; m2 = t; t = s1; s1 = s2; s2 = t; }
            if (m2 != -INFINITY) s1 += s2 * expf(m2 - m1);
            sm[tid] = m1; ss[tid] = s1;
        }
        __syncthreads();
    }
    if (tid == 0) part[blockIdx.x] = make_float2(sm[0], ss[0]);
}

__global__ void lse_p2_k(const float2* __restrict__ part, float* __restrict__ lse) {
    int tid = threadIdx.x;
    if (tid >= BEAM_) return;
    float m = -INFINITY, s = 0.0f;
    for (int c = 0; c < 32; ++c) {
        float2 p = part[tid * 32 + c];
        float m2 = p.x, s2 = p.y;
        if (m2 > m) { float t = m; m = m2; m2 = t; t = s; s = s2; s2 = t; }
        if (m2 != -INFINITY) s += s2 * expf(m2 - m);
    }
    lse[tid] = m + logf(s);
}

/* ------------------------------------------------------------------ */
/* 10. top-5 selection, two passes (deterministic tie-break: low idx)  */
/* ------------------------------------------------------------------ */
__global__ void topk_p1_k(const float* __restrict__ logits,
                          const int* __restrict__ st_cur,
                          const float* __restrict__ lse,
                          float* __restrict__ pv, int* __restrict__ pi) {
    __shared__ float sv[256];
    __shared__ int si[256], stt[256];
    const int tid = threadIdx.x;
    const int b = blockIdx.x / 40;
    const int c = blockIdx.x % 40;
    const int CH = (V_ + 39) / 40;           /* 1257 */
    const int v0 = c * CH, v1 = min(V_, v0 + CH);
    const float sc = ((const float*)st_cur)[ST_SC + b];
    const int fin = st_cur[ST_FIN + b];
    const float L = lse[b];
    float bv[5]; int bi[5];
#pragma unroll
    for (int i = 0; i < 5; ++i) { bv[i] = -INFINITY; bi[i] = 0x7fffffff; }
    for (int v = v0 + tid; v < v1; v += 256) {
        float x = fin ? (sc + ((v == EOS_) ? 0.0f : NEG_))
                      : (sc + logits[(long)b * VP_ + v] - L);
        int fid = b * V_ + v;
        if (x > bv[4] || (x == bv[4] && fid < bi[4])) {
            int p = 4;
            while (p > 0 && (x > bv[p - 1] || (x == bv[p - 1] && fid < bi[p - 1]))) {
                bv[p] = bv[p - 1]; bi[p] = bi[p - 1]; --p;
            }
            bv[p] = x; bi[p] = fid;
        }
    }
    int hp = 0;
    for (int r = 0; r < 5; ++r) {
        sv[tid] = (hp < 5) ? bv[hp] : -INFINITY;
        si[tid] = (hp < 5) ? bi[hp] : 0x7fffffff;
        stt[tid] = tid;
        __syncthreads();
        for (int off = 128; off > 0; off >>= 1) {
            if (tid < off) {
                float v2 = sv[tid + off]; int i2 = si[tid + off];
                if (v2 > sv[tid] || (v2 == sv[tid] && i2 < si[tid])) {
                    sv[tid] = v2; si[tid] = i2; stt[tid] = stt[tid + off];
                }
            }
            __syncthreads();
        }
        if (tid == 0) { pv[blockIdx.x * 5 + r] = sv[0]; pi[blockIdx.x * 5 + r] = si[0]; }
        if (tid == stt[0]) ++hp;
        __syncthreads();
    }
}

__global__ void topk_p2_k(const float* __restrict__ pv, const int* __restrict__ pi,
                          const float* __restrict__ h2buf,
                          float* __restrict__ hidden,
                          const int* __restrict__ st_old,
                          int* __restrict__ st_new, int tcur) {
    __shared__ float sval[1000];
    __shared__ int sidx[1000];
    __shared__ float rv[256];
    __shared__ int ri[256], rs[256];
    __shared__ float topv[5];
    __shared__ int topf[5], par[5], tk[5], pfin[5];
    const int tid = threadIdx.x;
    for (int i = tid; i < 1000; i += 256) { sval[i] = pv[i]; sidx[i] = pi[i]; }
    __syncthreads();
    for (int r = 0; r < 5; ++r) {
        float bvv = -INFINITY; int bii = 0x7fffffff, bsl = -1;
        for (int i = tid; i < 1000; i += 256) {
            float v = sval[i]; int ix = sidx[i];
            if (v > bvv || (v == bvv && ix < bii)) { bvv = v; bii = ix; bsl = i; }
        }
        rv[tid] = bvv; ri[tid] = bii; rs[tid] = bsl;
        __syncthreads();
        for (int off = 128; off > 0; off >>= 1) {
            if (tid < off) {
                float v2 = rv[tid + off]; int i2 = ri[tid + off];
                if (v2 > rv[tid] || (v2 == rv[tid] && i2 < ri[tid])) {
                    rv[tid] = v2; ri[tid] = i2; rs[tid] = rs[tid + off];
                }
            }
            __syncthreads();
        }
        if (tid == 0) {
            topv[r] = rv[0]; topf[r] = ri[0];
            if (rs[0] >= 0) sval[rs[0]] = -INFINITY;
        }
        __syncthreads();
    }
    if (tid < BEAM_) {
        int flat = topf[tid];
        int p = flat / V_, t_ = flat % V_;
        par[tid] = p; tk[tid] = t_;
        int pf = st_old[ST_FIN + p];
        pfin[tid] = pf;
        st_new[ST_TOK + tid] = t_;
        ((float*)st_new)[ST_SC + tid] = topv[tid];
        st_new[ST_FIN + tid] = (pf || t_ == EOS_) ? 1 : 0;
        st_new[ST_LEN + tid] = st_old[ST_LEN + p] + (pf ? 0 : 1);
    }
    __syncthreads();
    if (tid < BEAM_ * TMAX_) {
        int i = tid >> 5, u = tid & 31;
        int v = st_old[ST_SEQ + par[i] * TMAX_ + u];
        if (u == tcur) v = pfin[i] ? 0 : tk[i];
        st_new[ST_SEQ + i * TMAX_ + u] = v;
    }
    for (int idx = tid; idx < BEAM_ * H_; idx += 256) {
        int i = idx >> 9, j = idx & (H_ - 1);
        hidden[idx] = h2buf[(long)par[i] * H_ + j];
    }
}

/* ------------------------------------------------------------------ */
/* 11. final ranking + output (seqs as float, then norms)              */
/* ------------------------------------------------------------------ */
__global__ void out_k(const int* __restrict__ st, float* __restrict__ out) {
    if (threadIdx.x != 0) return;
    float norm[BEAM_]; int used[BEAM_];
    for (int i = 0; i < BEAM_; ++i) {
        norm[i] = ((const float*)st)[ST_SC + i] / (float)st[ST_LEN + i];
        used[i] = 0;
    }
    for (int r = 0; r < BEAM_; ++r) {
        int best = -1;
        for (int i = 0; i < BEAM_; ++i)
            if (!used[i] && (best < 0 || norm[i] > norm[best])) best = i;
        used[best] = 1;
        for (int u = 0; u < TMAX_; ++u)
            out[r * TMAX_ + u] = (float)st[ST_SEQ + best * TMAX_ + u];
        out[BEAM_ * TMAX_ + r] = norm[best];
    }
}

/* ================================================================== */
extern "C" void kernel_launch(void* const* d_in, const int* in_sizes, int n_in,
                              void* d_out, int out_size, void* d_ws, size_t ws_size,
                              hipStream_t stream) {
    const int*   toks    = (const int*)d_in[0];
    const float* emb     = (const float*)d_in[1];
    const float* eWih    = (const float*)d_in[2];
    const float* eWhh    = (const float*)d_in[3];
    const float* ebih    = (const float*)d_in[4];
    const float* ebhh    = (const float*)d_in[5];
    const float* dWih    = (const float*)d_in[6];
    const float* dWhh    = (const float*)d_in[7];
    const float* dbih    = (const float*)d_in[8];
    const float* dbhh    = (const float*)d_in[9];
    const float* Wc      = (const float*)d_in[10];
    const float* Wv      = (const float*)d_in[11];
    float* out = (float*)d_out;

    char* ws = (char*)d_ws;
    size_t off = 0;
    auto take = [&](size_t bytes) { size_t o = off; off = (off + bytes + 255) & ~(size_t)255; return o; };
    float*  src_emb = (float*)(ws + take((size_t)S_ * H_ * 4));
    float*  GI      = (float*)(ws + take((size_t)S_ * 3 * H_ * 4));
    float*  enc_out = (float*)(ws + take((size_t)S_ * H_ * 4));
    float*  hidden  = (float*)(ws + take((size_t)BEAM_ * H_ * 4));
    float*  h2buf   = (float*)(ws + take((size_t)BEAM_ * H_ * 4));
    float*  obuf    = (float*)(ws + take((size_t)BEAM_ * H_ * 4));
    bf16_t* Ap      = (bf16_t*)(ws + take((size_t)16 * 32 * 16 * 2));
    float*  logits  = (float*)(ws + take((size_t)BEAM_ * VP_ * 4));
    float2* lsep    = (float2*)(ws + take((size_t)160 * 8));
    float*  lse     = (float*)(ws + take((size_t)BEAM_ * 4));
    float*  tpv     = (float*)(ws + take((size_t)200 * 5 * 4));
    int*    tpi     = (int*)(ws + take((size_t)200 * 5 * 4));
    int*    st0     = (int*)(ws + take((size_t)ST_SIZE * 4));
    int*    st1     = (int*)(ws + take((size_t)ST_SIZE * 4));
    unsigned* ctr   = (unsigned*)(ws + take(4));
    bf16_t* Bp      = (bf16_t*)(ws + take((size_t)NT_ * 16 * 32 * 16 * 2));
    (void)ws_size; (void)in_sizes; (void)n_in; (void)out_size;

    /* encoder + one-time packing */
    enc_embed_k<<<(S_ * H_ + 255) / 256, 256, 0, stream>>>(toks, emb, src_emb, ctr);
    enc_gi_k<<<(S_ * 3 * H_ + 255) / 256, 256, 0, stream>>>(src_emb, eWih, ebih, GI);
    {
        long total = (long)NT_ * 16 * 32 * 16;
        pack_wvocab_k<<<(unsigned)((total + 255) / 256), 256, 0, stream>>>(Wv, Bp);
    }
    enc_scan_k<<<16, 256, 0, stream>>>(GI, eWhh, ebhh, enc_out, ctr);
    init_state_k<<<1, 256, 0, stream>>>(hidden, enc_out, st0, Ap);

    /* beam-search decode */
    for (int t = 0; t < TMAX_; ++t) {
        int* st_cur = (t & 1) ? st1 : st0;
        int* st_nxt = (t & 1) ? st0 : st1;
        dec_gru_k<<<BEAM_ * 2, 256, 0, stream>>>(st_cur, emb, hidden, dWih, dbih, dWhh, dbhh, h2buf);
        dec_attn_out_k<<<BEAM_, 256, 0, stream>>>(h2buf, enc_out, Wc, obuf, Ap);
        logits_wmma_k<<<(NT_ + 7) / 8, 256, 0, stream>>>(Ap, Bp, logits);
        lse_p1_k<<<BEAM_ * 32, 256, 0, stream>>>(logits, lsep);
        lse_p2_k<<<1, 32, 0, stream>>>(lsep, lse);
        topk_p1_k<<<BEAM_ * 40, 256, 0, stream>>>(logits, st_cur, lse, tpv, tpi);
        topk_p2_k<<<1, 256, 0, stream>>>(tpv, tpi, h2buf, hidden, st_cur, st_nxt, t);
    }
    /* after t=31, final state is in st0 (32 steps -> even buffer) */
    out_k<<<1, 32, 0, stream>>>(st0, out);
}
